// PerformerAttention_73624329388126
// MI455X (gfx1250) — matrix operand order repaired and checked by reading, not compiled, not run on hardware
//
#include <hip/hip_runtime.h>
#include <hip/hip_bf16.h>

// ---------------------------------------------------------------------------
// Performer attention on MI455X (gfx1250), wave32 + WMMA bf16 (f32 accum).
// All matmul operands are pre-converted / produced in bf16 so the WMMA hot
// loops contain no fp32->bf16 conversion VALU. B-panels are staged into LDS
// with gfx1250 async global->LDS copies (double buffered, ASYNCcnt).
//
//   q = Xq Wq^T + bq ; k ; v                         (bf16 WMMA GEMM, K=1024)
//   xn = normalize_per_head(q|k)                     (wave reduction)
//   xt = xn @ rm_h^T -> phi = s*[exp(xt),exp(-xt)]   (bf16 WMMA GEMM, K=64)
//   causal linear attention: chunked scan, chunk=64  (4 WMMA GEMMs per chunk,
//       fp32 state S^T(64x256)+z(256) in LDS, bf16 chunk tiles via async copy)
//   out = attn @ Wo^T + bo                           (bf16 WMMA GEMM, K=1024)
// ---------------------------------------------------------------------------

#define L_SEQ   2048
#define N_B     4
#define E_DIM   1024
#define H_N     16
#define HD      64
#define D_P     128
#define TWO_D   256
#define B_HEADS 64            // N_B * H_N
#define M_ROWS  8192          // L_SEQ * N_B
#define CHUNK   64

typedef __attribute__((ext_vector_type(16))) __bf16 v16bf;
typedef __attribute__((ext_vector_type(8)))  float  v8f;
typedef int v4i __attribute__((vector_size(16)));   // v4i32 for async builtins

union BF16x16 { v16bf v; unsigned short us[16]; };

__device__ __forceinline__ unsigned short f2bf(float f) {
  union { __bf16 h; unsigned short u; } c;
  c.h = (__bf16)f;                       // RNE, lowers to v_cvt_*bf16*
  return c.u;
}
__device__ __forceinline__ float bf2f(unsigned short h) {
  union { unsigned int u; float f; } c; c.u = ((unsigned int)h) << 16;
  return c.f;
}

__device__ __forceinline__ v8f wmma_bf16(v16bf a, v16bf b, v8f c) {
  // v_wmma_f32_16x16x32_bf16 : D = A(16x32) * B(32x16) + C(16x16 f32)
  return __builtin_amdgcn_wmma_f32_16x16x32_bf16(false, a, false, b,
                                                 (short)0, c, false, false);
}

// --- gfx1250 async global->LDS copy (16 bytes per lane), guarded ------------
__device__ __forceinline__ void async_copy16(const unsigned short* gsrc,
                                             unsigned short* ldst) {
#if __has_builtin(__builtin_amdgcn_global_load_async_to_lds_b128)
  __builtin_amdgcn_global_load_async_to_lds_b128(
      (__attribute__((address_space(1))) v4i*)gsrc,
      (__attribute__((address_space(3))) v4i*)ldst, 0, 0);
#else
  *(uint4*)ldst = *(const uint4*)gsrc;   // fallback: plain LDS store copy
#endif
}
template <int N>
__device__ __forceinline__ void async_wait_le() {
#if __has_builtin(__builtin_amdgcn_s_wait_asynccnt)
  __builtin_amdgcn_s_wait_asynccnt(N);   // requires constant integer
#endif
}

// --- WMMA fragment loaders (bf16 source, global or LDS) ---------------------
// A fragment (16x32 MxK): lane<16: M=lane, K=k0+[0..15]; lane>=16: K=k0+[16..31].
__device__ __forceinline__ v16bf load_a_bf(const unsigned short* A, int ld,
                                           int row0, int k0) {
  const int lane = threadIdx.x & 31;
  const unsigned short* p =
      A + (size_t)(row0 + (lane & 15)) * ld + k0 + ((lane >> 4) << 4);
  BF16x16 u;
#pragma unroll
  for (int i = 0; i < 16; ++i) u.us[i] = p[i];
  return u.v;
}
// B fragment (32x16 KxN) from an N x K row-major source (B[k][n] = W[n][k]).
__device__ __forceinline__ v16bf load_b_bf(const unsigned short* W, int ld,
                                           int n0, int k0) {
  const int lane = threadIdx.x & 31;
  const unsigned short* p =
      W + (size_t)(n0 + (lane & 15)) * ld + k0 + ((lane >> 4) << 4);
  BF16x16 u;
#pragma unroll
  for (int i = 0; i < 16; ++i) u.us[i] = p[i];
  return u.v;
}
// B fragment from fp32 LDS (used for the fp32 state S^T), converted per load.
__device__ __forceinline__ v16bf load_b_f32(const float* W, int ld,
                                            int n0, int k0) {
  const int lane = threadIdx.x & 31;
  const float* p =
      W + (size_t)(n0 + (lane & 15)) * ld + k0 + ((lane >> 4) << 4);
  BF16x16 u;
#pragma unroll
  for (int i = 0; i < 16; ++i) u.us[i] = f2bf(p[i]);
  return u.v;
}
// B fragment where the source is K x N row-major (strided): B[k][n]=S[k*ld+n].
__device__ __forceinline__ v16bf load_b_bf_strided(const unsigned short* S,
                                                   int ld, int n0, int k0) {
  const int lane = threadIdx.x & 31;
  const int n  = n0 + (lane & 15);
  const int kb = k0 + ((lane >> 4) << 4);
  BF16x16 u;
#pragma unroll
  for (int i = 0; i < 16; ++i) u.us[i] = S[(size_t)(kb + i) * ld + n];
  return u.v;
}

// ---------------------------------------------------------------------------
// fp32 -> bf16 conversion pass (bandwidth bound; 4 elements / thread).
// ---------------------------------------------------------------------------
__global__ void __launch_bounds__(256) cvt_f32_bf16(const float* __restrict__ src,
                                                    unsigned short* __restrict__ dst,
                                                    int n) {
  const int i = (int)(blockIdx.x * blockDim.x + threadIdx.x) * 4;
  if (i + 3 < n) {
#pragma unroll
    for (int j = 0; j < 4; ++j) dst[i + j] = f2bf(src[i + j]);
  }
}

// ---------------------------------------------------------------------------
// C = A(MxK,bf16) * W^T + bias ; W is N' x K bf16 row-major.
// Block = 256 threads = 8 waves; block tile 256(M) x 64(N).
// Wave tile 32x64: 2 A-frags x 4 B-frags = 8 WMMA per k-step (A and B panel
// reuse). The shared 64x32 B panel is double-buffered in LDS via async copies.
// ---------------------------------------------------------------------------
template <bool OUT_BF16>
__global__ void __launch_bounds__(256) gemm_bf16(const unsigned short* __restrict__ A,
                                                 const unsigned short* __restrict__ W,
                                                 const float* __restrict__ bias,
                                                 void* __restrict__ Cout,
                                                 int K, int ldc) {
  __shared__ unsigned short Bbuf[2][64][32];   // 2 x 4 KB
  const int tid  = (int)threadIdx.x;
  const int lane = tid & 31;
  const int wave = tid >> 5;
  const int m0 = (int)blockIdx.x * 256 + wave * 32;
  const int n0 = (int)blockIdx.y * 64;

  const int srow = tid >> 2;            // 0..63  (B-panel row n)
  const int scol = (tid & 3) << 3;      // 0,8,16,24 (halves)

  v8f acc[2][4] = {};
  // prologue stage of panel k0=0
  async_copy16(&W[(size_t)(n0 + srow) * K + scol], &Bbuf[0][srow][scol]);

  const int nk = K / 32;
  for (int it = 0; it < nk; ++it) {
    const int cur = it & 1;
    if (it + 1 < nk)
      async_copy16(&W[(size_t)(n0 + srow) * K + (it + 1) * 32 + scol],
                   &Bbuf[cur ^ 1][srow][scol]);
    if (it + 2 < nk)   // prefetch A stream two steps ahead
      __builtin_prefetch(&A[(size_t)(m0 + (lane & 15)) * K + (it + 2) * 32]);
    if (it + 1 < nk) async_wait_le<1>(); else async_wait_le<0>();
    __syncthreads();

    v16bf a0 = load_a_bf(A, K, m0, it * 32);
    v16bf a1 = load_a_bf(A, K, m0 + 16, it * 32);
#pragma unroll
    for (int j = 0; j < 4; ++j) {
      v16bf bb = load_b_bf(&Bbuf[cur][0][0], 32, j * 16, 0);
      acc[0][j] = wmma_bf16(a0, bb, acc[0][j]);
      acc[1][j] = wmma_bf16(a1, bb, acc[1][j]);
    }
    __syncthreads();
  }

  const int rowh = (lane >> 4) << 3;
  const int col  = lane & 15;
#pragma unroll
  for (int mi = 0; mi < 2; ++mi) {
#pragma unroll
    for (int j = 0; j < 4; ++j) {
      const float bj = bias[n0 + j * 16 + col];
#pragma unroll
      for (int r = 0; r < 8; ++r) {
        const size_t idx =
            (size_t)(m0 + mi * 16 + rowh + r) * ldc + n0 + j * 16 + col;
        const float v = acc[mi][j][r] + bj;
        if constexpr (OUT_BF16) ((unsigned short*)Cout)[idx] = f2bf(v);
        else                    ((float*)Cout)[idx] = v;
      }
    }
  }
}

// ---------------------------------------------------------------------------
// Per-(l,n,h) row L2-normalization (bf16 in/out); head-major xn (H, M, 64).
// ---------------------------------------------------------------------------
__global__ void __launch_bounds__(256) norm_heads(const unsigned short* __restrict__ X,
                                                  unsigned short* __restrict__ XN) {
  const int gw   = (int)((blockIdx.x * blockDim.x + threadIdx.x) >> 5);
  const int lane = (int)threadIdx.x & 31;
  const int m = gw >> 4;   // row in (L*N)
  const int h = gw & 15;   // head
  const unsigned short* p = X + (size_t)m * E_DIM + h * HD;
  const float x0 = bf2f(p[lane]), x1 = bf2f(p[lane + 32]);
  float ss = x0 * x0 + x1 * x1;
#pragma unroll
  for (int off = 16; off >= 1; off >>= 1) ss += __shfl_xor(ss, off, 32);
  const float inv = 1.0f / fmaxf(sqrtf(ss), 1e-12f);
  unsigned short* q = XN + ((size_t)h * M_ROWS + m) * HD;
  q[lane]      = f2bf(x0 * inv);
  q[lane + 32] = f2bf(x1 * inv);
}

// ---------------------------------------------------------------------------
// xt = xn_h (8192x64) @ rm_h^T (64x128) -> phi = (1/16)*[exp(xt), exp(-xt)]
// phi written bf16 in the reference's plain-reshape (64,2048,256) layout:
// flat offset (m*16 + h)*256 + d, m = l*N+n.
// ---------------------------------------------------------------------------
__global__ void __launch_bounds__(128) phi_gemm(const unsigned short* __restrict__ XN,
                                                const unsigned short* __restrict__ RM,
                                                unsigned short* __restrict__ PHI) {
  const int h = (int)blockIdx.z;
  const unsigned short* A = XN + (size_t)h * M_ROWS * HD;
  const unsigned short* W = RM + (size_t)h * D_P * HD;
  const int lane = (int)threadIdx.x & 31;
  const int wave = (int)threadIdx.x >> 5;
  const int m0 = (int)blockIdx.x * 64 + wave * 16;
  const int n0 = (int)blockIdx.y * 64;
  v8f acc[4] = {};
  for (int k0 = 0; k0 < HD; k0 += 32) {
    v16bf a = load_a_bf(A, HD, m0, k0);
#pragma unroll
    for (int j = 0; j < 4; ++j) {
      v16bf bb = load_b_bf(W, HD, n0 + j * 16, k0);
      acc[j] = wmma_bf16(a, bb, acc[j]);
    }
  }
  const int rowh = (lane >> 4) << 3;
  const int col  = lane & 15;
  const float scale = 0.0625f;  // 1/sqrt(2*D) = 1/16
#pragma unroll
  for (int j = 0; j < 4; ++j) {
#pragma unroll
    for (int r = 0; r < 8; ++r) {
      const int m = m0 + rowh + r;
      const int d = n0 + j * 16 + col;
      const size_t base = ((size_t)m * H_N + h) * TWO_D;
      const float xt = acc[j][r];
      PHI[base + d]       = f2bf(scale * expf(xt));
      PHI[base + D_P + d] = f2bf(scale * expf(-xt));
    }
  }
}

// ---------------------------------------------------------------------------
// Chunked causal linear attention. One workgroup (256 thr = 8 waves) per head.
// fp32 state S^T (64x256) + z (256) persist in LDS across 32 chunks of 64.
// Per chunk (inclusive causal mask j<=i):
//   Am  = mask(Qc Kc^T)                 den += rowsum(Am) + Qc . z_old
//   out = (Qc S_old^T + Am Vc) / (den+eps)        -> bf16 attn (L,N,E layout)
//   S^T += Vc^T Kc ;  z += colsum(Kc)
// Qc/Kc tiles staged global->LDS with async b128 copies.
// ---------------------------------------------------------------------------
__global__ void __launch_bounds__(256) causal_scan(const unsigned short* __restrict__ PHIQ,
                                                   const unsigned short* __restrict__ PHIK,
                                                   const unsigned short* __restrict__ V,
                                                   unsigned short* __restrict__ OUT) {
  __shared__ float S_T[HD][TWO_D];            // 64 KB, S transposed: [hd][d]
  __shared__ float zvec[TWO_D];
  __shared__ float den[CHUNK];
  __shared__ unsigned short Qc[CHUNK][TWO_D]; // 32 KB bf16
  __shared__ unsigned short Kc[CHUNK][TWO_D]; // 32 KB bf16
  __shared__ unsigned short VT[HD][CHUNK];    //  8 KB bf16, V chunk transposed
  __shared__ unsigned short Am[CHUNK][CHUNK]; //  8 KB bf16, masked Q K^T

  const int b    = (int)blockIdx.x;    // scrambled-phi batch index
  const int tid  = (int)threadIdx.x;
  const int lane = tid & 31;
  const int wave = tid >> 5;
  const int nb = b >> 4;               // batch n (v/out use b = n*H + h)
  const int hb = b & 15;               // head h

  for (int i = tid; i < HD * TWO_D; i += 256) (&S_T[0][0])[i] = 0.f;
  if (tid < TWO_D) zvec[tid] = 0.f;
  __syncthreads();

  const unsigned short* phiq_b = PHIQ + (size_t)b * L_SEQ * TWO_D;
  const unsigned short* phik_b = PHIK + (size_t)b * L_SEQ * TWO_D;

  for (int c0 = 0; c0 < L_SEQ; c0 += CHUNK) {
    // ---- async-stage Qc/Kc (bf16, straight copy), manual transpose for VT --
    for (int i = tid; i < (CHUNK * TWO_D) / 8; i += 256) {  // 16B granules
      const int j = i >> 5;            // chunk row 0..63
      const int d = (i & 31) << 3;     // feature 0..248 step 8
      async_copy16(&phiq_b[(size_t)(c0 + j) * TWO_D + d], &Qc[j][d]);
      async_copy16(&phik_b[(size_t)(c0 + j) * TWO_D + d], &Kc[j][d]);
    }
    for (int i = tid; i < CHUNK * HD; i += 256) {
      const int j = i >> 6, hd = i & 63;
      VT[hd][j] = V[(size_t)((c0 + j) * N_B + nb) * E_DIM + hb * HD + hd];
    }
    if (tid < CHUNK) den[tid] = 0.f;
    async_wait_le<0>();
    __syncthreads();

    // ---- den[m] += phi_q[m] . z_old (4 partial sums per row) ----
    {
      const int m = tid >> 2, qq = tid & 3;
      float s = 0.f;
      for (int d = qq * 64; d < qq * 64 + 64; ++d) s += bf2f(Qc[m][d]) * zvec[d];
      atomicAdd(&den[m], s);
    }

    // ---- Am = mask(Qc Kc^T); den += rowsum ---- (16 tiles / 8 waves)
    for (int t = wave * 2; t < wave * 2 + 2; ++t) {
      const int ti = t >> 2, tj = t & 3;
      v8f acc = {};
      for (int k0 = 0; k0 < TWO_D; k0 += 32) {
        v16bf a  = load_a_bf(&Qc[0][0], TWO_D, ti * 16, k0);
        v16bf bb = load_b_bf(&Kc[0][0], TWO_D, tj * 16, k0);
        acc = wmma_bf16(a, bb, acc);
      }
      const int rowh = (lane >> 4) << 3;
      const int col  = lane & 15;
#pragma unroll
      for (int r = 0; r < 8; ++r) {
        const int i = ti * 16 + rowh + r;
        const int j = tj * 16 + col;
        const float val = (j <= i) ? acc[r] : 0.f;
        Am[i][j] = f2bf(val);
        float rs = val;                 // reduce over the 16 j-lanes
        rs += __shfl_xor(rs, 1, 32);
        rs += __shfl_xor(rs, 2, 32);
        rs += __shfl_xor(rs, 4, 32);
        rs += __shfl_xor(rs, 8, 32);
        if (col == 0) atomicAdd(&den[i], rs);
      }
    }
    __syncthreads();

    // ---- out = (Qc S_old^T + Am Vc) / (den + eps) ----
    for (int t = wave * 2; t < wave * 2 + 2; ++t) {
      const int ti = t >> 2, tn = t & 3;
      v8f acc = {};
      for (int k0 = 0; k0 < TWO_D; k0 += 32) {            // inter-chunk: Q @ S
        v16bf a  = load_a_bf(&Qc[0][0], TWO_D, ti * 16, k0);
        v16bf bb = load_b_f32(&S_T[0][0], TWO_D, tn * 16, k0);
        acc = wmma_bf16(a, bb, acc);
      }
      for (int k0 = 0; k0 < CHUNK; k0 += 32) {            // intra-chunk: A @ V
        v16bf a  = load_a_bf(&Am[0][0], CHUNK, ti * 16, k0);
        v16bf bb = load_b_bf(&VT[0][0], CHUNK, tn * 16, k0);
        acc = wmma_bf16(a, bb, acc);
      }
      const int rowh = (lane >> 4) << 3;
      const int col  = lane & 15;
#pragma unroll
      for (int r = 0; r < 8; ++r) {
        const int i  = ti * 16 + rowh + r;
        const int hd = tn * 16 + col;
        const int l  = c0 + i;
        OUT[(size_t)(l * N_B + nb) * E_DIM + hb * HD + hd] =
            f2bf(acc[r] / (den[i] + 1e-6f));
      }
    }
    __syncthreads();

    // ---- S^T += Vc^T Kc (64 tiles / 8 waves); z += colsum(Kc) ----
    for (int t = wave * 8; t < wave * 8 + 8; ++t) {
      const int tm = t >> 4, tn = t & 15;
      const int rowh = (lane >> 4) << 3;
      const int col  = lane & 15;
      v8f acc;
#pragma unroll
      for (int r = 0; r < 8; ++r) acc[r] = S_T[tm * 16 + rowh + r][tn * 16 + col];
      for (int k0 = 0; k0 < CHUNK; k0 += 32) {
        v16bf a  = load_a_bf(&VT[0][0], CHUNK, tm * 16, k0);
        v16bf bb = load_b_bf_strided(&Kc[0][0], TWO_D, tn * 16, k0);
        acc = wmma_bf16(a, bb, acc);
      }
#pragma unroll
      for (int r = 0; r < 8; ++r) S_T[tm * 16 + rowh + r][tn * 16 + col] = acc[r];
    }
    {
      const int d = tid;               // one thread per feature
      float s = 0.f;
      for (int j = 0; j < CHUNK; ++j) s += bf2f(Kc[j][d]);
      zvec[d] += s;
    }
    __syncthreads();
  }
}

// ---------------------------------------------------------------------------
extern "C" void kernel_launch(void* const* d_in, const int* in_sizes, int n_in,
                              void* d_out, int out_size, void* d_ws, size_t ws_size,
                              hipStream_t stream) {
  (void)in_sizes; (void)n_in; (void)out_size; (void)ws_size;
  const float* query = (const float*)d_in[0];
  const float* key   = (const float*)d_in[1];
  const float* value = (const float*)d_in[2];
  const float* wq = (const float*)d_in[3];
  const float* bq = (const float*)d_in[4];
  const float* wk = (const float*)d_in[5];
  const float* bk = (const float*)d_in[6];
  const float* wv = (const float*)d_in[7];
  const float* bv = (const float*)d_in[8];
  const float* wo = (const float*)d_in[9];
  const float* bo = (const float*)d_in[10];
  const float* rm = (const float*)d_in[11];
  float* out = (float*)d_out;

  // workspace layout (bf16 halves); total ~294 MB
  unsigned short* ws = (unsigned short*)d_ws;
  const size_t ME  = (size_t)M_ROWS * E_DIM;            // 8,388,608
  const size_t EE  = (size_t)E_DIM * E_DIM;             // 1,048,576
  const size_t RMS = (size_t)H_N * D_P * HD;            //   131,072
  const size_t PHI = (size_t)B_HEADS * L_SEQ * TWO_D;   // 33,554,432
  unsigned short* qry_bf = ws;
  unsigned short* key_bf = qry_bf + ME;
  unsigned short* val_bf = key_bf + ME;
  unsigned short* wq_bf  = val_bf + ME;
  unsigned short* wk_bf  = wq_bf + EE;
  unsigned short* wv_bf  = wk_bf + EE;
  unsigned short* wo_bf  = wv_bf + EE;
  unsigned short* rm_bf  = wo_bf + EE;
  unsigned short* qbf    = rm_bf + RMS;
  unsigned short* kbf    = qbf + ME;
  unsigned short* vbf    = kbf + ME;
  unsigned short* xnq    = vbf + ME;
  unsigned short* xnk    = xnq + ME;
  unsigned short* phiq   = xnk + ME;
  unsigned short* phik   = phiq + PHI;
  unsigned short* attnbf = phik + PHI;

  // 1) fp32 -> bf16 operand conversion (one pass each)
  cvt_f32_bf16<<<ME / 1024, 256, 0, stream>>>(query, qry_bf, (int)ME);
  cvt_f32_bf16<<<ME / 1024, 256, 0, stream>>>(key,   key_bf, (int)ME);
  cvt_f32_bf16<<<ME / 1024, 256, 0, stream>>>(value, val_bf, (int)ME);
  cvt_f32_bf16<<<EE / 1024, 256, 0, stream>>>(wq, wq_bf, (int)EE);
  cvt_f32_bf16<<<EE / 1024, 256, 0, stream>>>(wk, wk_bf, (int)EE);
  cvt_f32_bf16<<<EE / 1024, 256, 0, stream>>>(wv, wv_bf, (int)EE);
  cvt_f32_bf16<<<EE / 1024, 256, 0, stream>>>(wo, wo_bf, (int)EE);
  cvt_f32_bf16<<<RMS / 1024, 256, 0, stream>>>(rm, rm_bf, (int)RMS);

  // 2) input projections (bf16 out); block tile 256x64, wave tile 32x64
  const dim3 blkG(256);
  const dim3 gG(M_ROWS / 256, E_DIM / 64);
  gemm_bf16<true><<<gG, blkG, 0, stream>>>(qry_bf, wq_bf, bq, qbf, E_DIM, E_DIM);
  gemm_bf16<true><<<gG, blkG, 0, stream>>>(key_bf, wk_bf, bk, kbf, E_DIM, E_DIM);
  gemm_bf16<true><<<gG, blkG, 0, stream>>>(val_bf, wv_bf, bv, vbf, E_DIM, E_DIM);

  // 3) per-head normalization
  norm_heads<<<(M_ROWS * H_N) / 8, 256, 0, stream>>>(qbf, xnq);
  norm_heads<<<(M_ROWS * H_N) / 8, 256, 0, stream>>>(kbf, xnk);

  // 4) random-feature maps
  const dim3 pG(M_ROWS / 64, D_P / 64, H_N);
  phi_gemm<<<pG, 128, 0, stream>>>(xnq, rm_bf, phiq);
  phi_gemm<<<pG, 128, 0, stream>>>(xnk, rm_bf, phik);

  // 5) chunked causal linear attention
  causal_scan<<<B_HEADS, 256, 0, stream>>>(phiq, phik, vbf, attnbf);

  // 6) output projection (fp32 out + bias)
  gemm_bf16<false><<<gG, blkG, 0, stream>>>(attnbf, wo_bf, bo, out, E_DIM, E_DIM);
}
